// QuantumDMRGLayer_46969762349702
// MI455X (gfx1250) — matrix operand order repaired and checked by reading, not compiled, not run on hardware
//
#include <hip/hip_runtime.h>

typedef __attribute__((ext_vector_type(16))) _Float16 v16h;
typedef __attribute__((ext_vector_type(8)))  float    v8f;

#define DIMVEC 784
#define POSIDX 392
#define MDIM   16
#define NLAB   10
#define BATCH  256
#define LSTEPS (POSIDX - 1)          // 391
#define RSTEPS (DIMVEC - 2 - POSIDX) // 390
#define SEGS   16
#define SEGLEN 25                    // ceil(391/16)

// ---- workspace layout (bytes) ----
// Bleft : f16 [LSTEPS][32][16]  B-operand rows (lane = K = p*16+i, cols j)
// Bright: f16 [RSTEPS][32][16]  transposed rows (lane = K = p*16+j, cols i)
// Apos  : f16 [512][16]         rows K = p*256+i*16+j, cols n (zero-pad 10->16)
// P     : f16 [2][SEGS][256][16][16]  per-batch segment products (4 MB)
#define WS_BLEFT   0
#define WS_BRIGHT  (LSTEPS * 32 * 16 * 2)               // 400384
#define WS_APOS    (WS_BRIGHT + RSTEPS * 32 * 16 * 2)   // 799744
#define WS_P       (WS_APOS + 512 * 16 * 2 + 1024)      // 817152 (aligned)

// wave-local LDS fence: same-wave DS ops complete in order; the dscnt wait +
// compiler memory clobber makes cross-lane LDS reuse safe without s_barrier.
#define WAVE_LDS_FENCE() asm volatile("s_wait_dscnt 0x0" ::: "memory")

// ---------------- prep: pack site tensors into f16 B-operand layout ----------------
__global__ void dmrg_prep_kernel(const float* __restrict__ A_left,
                                 const float* __restrict__ A_right,
                                 const float* __restrict__ A_pos,
                                 _Float16* __restrict__ ws)
{
    const int gid = blockIdx.x * blockDim.x + threadIdx.x;
    const int NLR = LSTEPS * 32;   // 12512 rows
    const int NRR = RSTEPS * 32;   // 12480 rows
    _Float16* bleft  = ws + WS_BLEFT / 2;
    _Float16* bright = ws + WS_BRIGHT / 2;
    _Float16* apos   = ws + WS_APOS / 2;

    if (gid < NLR) {
        const float* src = A_left + (size_t)gid * 16;       // row k*32 + (p*16+i)
        _Float16*    dst = bleft + (size_t)gid * 16;
        for (int j = 0; j < 16; ++j) dst[j] = (_Float16)src[j];
    } else if (gid < NLR + NRR) {
        const int q  = gid - NLR;                           // row r*32 + (p*16+j)
        const int r  = q >> 5;
        const int l  = q & 31;
        const int p  = l >> 4;
        const int jc = l & 15;
        const float* src = A_right + (size_t)r * 512 + p * 256 + jc;
        _Float16*    dst = bright + (size_t)q * 16;
        for (int i = 0; i < 16; ++i) dst[i] = (_Float16)src[i * 16];
    } else if (gid < NLR + NRR + 512) {
        const int K = gid - NLR - NRR;
        const float* src = A_pos + (size_t)K * 10;
        _Float16*    dst = apos + (size_t)K * 16;
        for (int n = 0; n < 16; ++n)
            dst[n] = (n < NLAB) ? (_Float16)src[n] : (_Float16)0.0f;
    }
}

// ---------------- segments: blocked scan over per-batch 16x16 matrix products ----------------
// 8192 waves: wave w -> side = w>>12, seg s = (w>>8)&15, batch b = w&255.
// Each wave: P = identity, then P <- (x-scaled P) x Bsite via one WMMA per site.
__global__ void dmrg_seg_kernel(const float* __restrict__ x,
                                const _Float16* __restrict__ ws_h,
                                _Float16* __restrict__ ws_p)
{
    __shared__ float V[8][2][16][20];   // per-wave slab, padded rows

    const int lane  = threadIdx.x & 31;
    const int wslab = threadIdx.x >> 5;
    const int m     = lane & 15;
    const int half  = lane >> 4;

    // wave-uniform ids made explicit scalars -> scalar loop control, EXEC stays full
    const int wu   = __builtin_amdgcn_readfirstlane(blockIdx.x * 8 + wslab);
    const int side = wu >> 12;
    const int s    = (wu >> 8) & 15;
    const int b    = wu & 255;

    const int nsteps = side ? RSTEPS : LSTEPS;
    const int start  = s * SEGLEN;
    const int end    = (start + SEGLEN < nsteps) ? (start + SEGLEN) : nsteps;
    const int len    = end - start;
    const _Float16* __restrict__ bbase = ws_h + (side ? WS_BRIGHT / 2 : WS_BLEFT / 2);
    const float* __restrict__ xb = x + (size_t)b * DIMVEC * 2;

    // preload this segment's x pair per lane (lane u <-> step start+u); broadcast later.
    const int colL  = side ? (DIMVEC - 2 - start - lane) : (1 + start + lane);
    const float xv0 = xb[colL * 2 + 0];
    const float xv1 = xb[colL * 2 + 1];

    // identity init
    #pragma unroll
    for (int c = 0; c < 8; ++c) {
        const int j = half * 8 + c;
        V[wslab][0][m][j] = (m == j) ? 1.0f : 0.0f;
    }
    WAVE_LDS_FENCE();

    // software-pipelined B-operand load
    const int k0 = side ? (RSTEPS - 1 - start) : start;
    v16h bm;
    for (int e = 0; e < 16; ++e) bm[e] = bbase[(size_t)k0 * 512 + (((e >> 3) << 4) + half * 8 + (e & 7)) * 16 + m];

    v8f d = {};
    for (int u = 0; u < len; ++u) {
        const int pb = u & 1;
        const float x0 = __int_as_float(__builtin_amdgcn_readlane(__float_as_int(xv0), u));
        const float x1 = __int_as_float(__builtin_amdgcn_readlane(__float_as_int(xv1), u));

        const float4 r0 = *(const float4*)&V[wslab][pb][m][half * 8];
        const float4 r1 = *(const float4*)&V[wslab][pb][m][half * 8 + 4];

        v16h a;
        a[0]  = (_Float16)(x0 * r0.x);  a[1]  = (_Float16)(x0 * r0.y);
        a[2]  = (_Float16)(x0 * r0.z);  a[3]  = (_Float16)(x0 * r0.w);
        a[4]  = (_Float16)(x0 * r1.x);  a[5]  = (_Float16)(x0 * r1.y);
        a[6]  = (_Float16)(x0 * r1.z);  a[7]  = (_Float16)(x0 * r1.w);
        a[8]  = (_Float16)(x1 * r0.x);  a[9]  = (_Float16)(x1 * r0.y);
        a[10] = (_Float16)(x1 * r0.z);  a[11] = (_Float16)(x1 * r0.w);
        a[12] = (_Float16)(x1 * r1.x);  a[13] = (_Float16)(x1 * r1.y);
        a[14] = (_Float16)(x1 * r1.z);  a[15] = (_Float16)(x1 * r1.w);

        // prefetch next site's B rows before the WMMA (overlaps its latency)
        const int un = (u + 1 < len) ? (u + 1) : u;
        const int tn = start + un;
        const int kn = side ? (RSTEPS - 1 - tn) : tn;
        v16h bmn;
        for (int e = 0; e < 16; ++e) bmn[e] = bbase[(size_t)kn * 512 + (((e >> 3) << 4) + half * 8 + (e & 7)) * 16 + m];

        v8f cz = {};
        d = __builtin_amdgcn_wmma_f32_16x16x32_f16(false, a, false, bm,
                                                   (short)0, cz, false, false);

        const int nb = 1 - pb;
        #pragma unroll
        for (int v = 0; v < 8; ++v)
            V[wslab][nb][v + 8 * half][m] = d[v];
        bm = bmn;
        WAVE_LDS_FENCE();
    }

    // store segment product row-major f16: P[side][s][b][i][j]
    _Float16* __restrict__ dst = ws_p + ((size_t)((side * SEGS + s) * 256 + b)) * 256;
    #pragma unroll
    for (int v = 0; v < 8; ++v)
        dst[(v + 8 * half) * 16 + m] = (_Float16)d[v];
}

// ---------------- tail: combine (both sides) + final A_pos contraction, fused ----------------
// 16 blocks x 512 threads. Phase 1: 2 sides x 16 batch-rows x 16 cols apply the 16
// per-batch segment matrices (VALU, tiny). Phase 2: wave 0 does the K=512 WMMA GEMM.
__global__ void dmrg_tail_kernel(const float* __restrict__ x,
                                 const float* __restrict__ A0,
                                 const float* __restrict__ A_last,
                                 const _Float16* __restrict__ ws_h,
                                 const _Float16* __restrict__ ws_p,
                                 float* __restrict__ out)
{
    __shared__ float V[2][2][16][17];   // [side][buf][bl][j]
    __shared__ float sv[2][16][16];     // final vL / vR for this tile

    const int tid  = threadIdx.x;       // 0..511
    const int side = tid >> 8;
    const int t8   = tid & 255;
    const int j    = t8 & 15;
    const int bl   = t8 >> 4;
    const int tile = blockIdx.x;
    const int b    = tile * 16 + bl;

    // boundary vector
    {
        const float* __restrict__ Amat = side ? A_last : A0;
        const int xcol0 = side ? (DIMVEC - 1) : 0;
        const float xa = x[((size_t)b * DIMVEC + xcol0) * 2 + 0];
        const float xb = x[((size_t)b * DIMVEC + xcol0) * 2 + 1];
        V[side][0][bl][j] = xa * Amat[j] + xb * Amat[16 + j];
    }
    __syncthreads();

    float acc = 0.0f;
    for (int s = 0; s < SEGS; ++s) {
        const int cur = s & 1;
        const _Float16* __restrict__ P =
            ws_p + ((size_t)((side * SEGS + s) * 256 + b)) * 256;
        acc = 0.0f;
        #pragma unroll
        for (int i = 0; i < 16; ++i)
            acc += V[side][cur][bl][i] * (float)P[i * 16 + j];
        V[side][1 - cur][bl][j] = acc;
        __syncthreads();
    }
    sv[side][bl][j] = acc;
    __syncthreads();

    // ---- phase 2: wave 0 only (full 32-lane EXEC for WMMA) ----
    if (tid < 32) {
        const int lane = tid;
        const int m    = lane & 15;
        const int half = lane >> 4;
        const int bb   = tile * 16 + m;
        const _Float16* __restrict__ apos = ws_h + WS_APOS / 2;

        const float x0 = x[((size_t)bb * DIMVEC + POSIDX) * 2 + 0];
        const float x1 = x[((size_t)bb * DIMVEC + POSIDX) * 2 + 1];

        float tv0[16], tv1[16];
        #pragma unroll
        for (int i = 0; i < 16; ++i) {
            const float v = sv[0][m][i];
            tv0[i] = x0 * v;
            tv1[i] = x1 * v;
        }
        float vr8[8];
        #pragma unroll
        for (int c = 0; c < 8; ++c) vr8[c] = sv[1][m][half * 8 + c];

        v8f acc2 = {};
        #pragma unroll
        for (int c = 0; c < 16; ++c) {
            const int i1 = (2 * c) & 15;
            const int i2 = i1 + 1;
            const float ta = (c >= 8) ? tv1[i1] : tv0[i1];
            const float tb = (c >= 8) ? tv1[i2] : tv0[i2];
            v16h a;
            #pragma unroll
            for (int u = 0; u < 8; ++u) {
                a[u]     = (_Float16)(ta * vr8[u]);
                a[8 + u] = (_Float16)(tb * vr8[u]);
            }
            v16h bm;
            for (int e = 0; e < 16; ++e) bm[e] = apos[(size_t)(c * 32 + ((e >> 3) << 4) + half * 8 + (e & 7)) * 16 + m];
            acc2 = __builtin_amdgcn_wmma_f32_16x16x32_f16(false, a, false, bm,
                                                          (short)0, acc2, false, false);
        }

        if (m < NLAB) {
            #pragma unroll
            for (int v = 0; v < 8; ++v)
                out[(size_t)(tile * 16 + v + 8 * half) * NLAB + m] = acc2[v];
        }
    }
}

extern "C" void kernel_launch(void* const* d_in, const int* in_sizes, int n_in,
                              void* d_out, int out_size, void* d_ws, size_t ws_size,
                              hipStream_t stream)
{
    (void)in_sizes; (void)n_in; (void)out_size; (void)ws_size;
    const float* x       = (const float*)d_in[0];
    const float* A0      = (const float*)d_in[1];
    const float* A_left  = (const float*)d_in[2];
    const float* A_pos   = (const float*)d_in[3];
    const float* A_right = (const float*)d_in[4];
    const float* A_last  = (const float*)d_in[5];
    float*       out     = (float*)d_out;
    _Float16*    wsh     = (_Float16*)d_ws;
    _Float16*    wsp     = (_Float16*)((char*)d_ws + WS_P);

    const int prep_rows = LSTEPS * 32 + RSTEPS * 32 + 512;   // 25504
    dmrg_prep_kernel<<<(prep_rows + 255) / 256, 256, 0, stream>>>(A_left, A_right, A_pos, wsh);
    dmrg_seg_kernel<<<1024, 256, 0, stream>>>(x, wsh, wsp);          // 8192 waves
    dmrg_tail_kernel<<<16, 512, 0, stream>>>(x, A0, A_last, wsh, wsp, out);
}
